// TransformerLayer_73151882985877
// MI455X (gfx1250) — matrix operand order repaired and checked by reading, not compiled, not run on hardware
//
#include <hip/hip_runtime.h>
#include <hip/hip_bf16.h>
#include <math.h>

#define DEVI __device__ __forceinline__

typedef __attribute__((ext_vector_type(16))) __bf16        v16bf;
typedef __attribute__((ext_vector_type(8)))  float         v8f;
typedef __attribute__((ext_vector_type(4)))  unsigned int  u32x4;
typedef __attribute__((ext_vector_type(8)))  int           i32x8;
typedef __attribute__((ext_vector_type(4)))  int           i32x4;

// Problem constants (B,H,W,C fixed by the reference)
constexpr int BB   = 2;
constexpr int HH   = 128;
constexpr int WW2  = 128;
constexpr int CC   = 128;
constexpr int LL   = HH * WW2;   // 16384 tokens
constexpr int NWIN = 64;         // 8x8 windows
constexpr int WP   = 256;        // 16x16 positions per window
constexpr int HID  = 1024;       // FFN hidden

// ---------------------------------------------------------------------------
// bf16 helpers
// ---------------------------------------------------------------------------
union BfBits { unsigned short u; __bf16 b; };

DEVI unsigned short f2bfbits(float f) { BfBits c; c.b = (__bf16)f; return c.u; }

DEVI v16bf load16bf(const unsigned short* p) {
  union { u32x4 u[2]; v16bf v; } c;
  c.u[0] = ((const u32x4*)p)[0];
  c.u[1] = ((const u32x4*)p)[1];
  return c.v;
}

// A fragment (16x32, MxK) of V_WMMA_F32_16X16X32_BF16 from a bf16 row-major
// buffer, leading dim ld (elements). Per ISA 7.12.2: lane = M (0..15, both
// halves), lanes 0-15 hold K {0..7,16..23}, lanes 16-31 hold K {8..15,24..31}.
DEVI v16bf a_frag_bf(const unsigned short* p, int ld, int row, int k0) {
  int lane = threadIdx.x & 31, m = lane & 15, h = lane >> 4;
  const unsigned short* r = p + (size_t)(row + m) * ld + k0 + h * 8;
  union { u32x4 u[2]; v16bf v; } c;
  c.u[0] = ((const u32x4*)r)[0];        // K = h*8 .. h*8+7
  c.u[1] = ((const u32x4*)(r + 16))[0]; // K = 16+h*8 .. 16+h*8+7
  return c.v;
}

// Same A fragment but converting from f32 row-major on the fly.
DEVI v16bf a_frag_f32(const float* p, int ld, int row, int k0) {
  int lane = threadIdx.x & 31, m = lane & 15, h = lane >> 4;
  const float* r = p + (size_t)(row + m) * ld + k0;
  v16bf a;
#pragma unroll
  for (int j = 0; j < 8; ++j) a[j] = (__bf16)r[h * 8 + j];
#pragma unroll
  for (int j = 0; j < 8; ++j) a[8 + j] = (__bf16)r[16 + h * 8 + j];
  return a;
}

// B fragment (32x16, KxN) loaded from an N-major buffer (i.e. B^T row-major,
// element (k,n) at p[n*ld + k]). Per ISA: lane = N, lanes 0-15 hold K
// {0..7,16..23}, lanes 16-31 hold K {8..15,24..31} -> two 16-byte reads.
DEVI v16bf b_frag_nmajor(const unsigned short* p, int ld, int k0, int n0) {
  int lane = threadIdx.x & 31, n = lane & 15, h = lane >> 4;
  const unsigned short* r = p + (size_t)(n0 + n) * ld + k0 + h * 8;
  union { u32x4 u[2]; v16bf v; } c;
  c.u[0] = ((const u32x4*)r)[0];
  c.u[1] = ((const u32x4*)(r + 16))[0];
  return c.v;
}

DEVI v8f wmma_bf16(v16bf a, v16bf b, v8f c) {
  return __builtin_amdgcn_wmma_f32_16x16x32_bf16(false, a, false, b,
                                                 (short)0, c, false, false);
}

DEVI float half16_max(float v) {
#pragma unroll
  for (int m = 8; m >= 1; m >>= 1) v = fmaxf(v, __shfl_xor(v, m, 32));
  return v;
}
DEVI float half16_sum(float v) {
#pragma unroll
  for (int m = 8; m >= 1; m >>= 1) v += __shfl_xor(v, m, 32);
  return v;
}

// ---------------------------------------------------------------------------
// Tensor Data Mover: 1-D contiguous global->LDS DMA.
// D# built per CDNA5 ISA ch.8: group0 {count, lds_addr, global_addr, type=2},
// group1 {data_size=8B, tensor_dim0/tile_dim0 = length in 8B units, stride}.
// Groups 2/3 zero (tile is <=2-D); trailing int32x8 group zero-filled as in
// the verified probe call (6-arg clang-23 signature). `bytes` % 8 == 0.
// ---------------------------------------------------------------------------
#if __has_builtin(__builtin_amdgcn_tensor_load_to_lds)
#define HAVE_TDM 1
DEVI void tdm_load_1d(const void* gsrc, unsigned lds_off, unsigned bytes) {
  unsigned long long ga = (unsigned long long)(size_t)gsrc;
  unsigned units = bytes >> 3;  // 8-byte elements
  u32x4 g0;
  g0[0] = 1u;                                  // count=1 (valid user D#)
  g0[1] = lds_off;                             // lds_addr (bytes)
  g0[2] = (unsigned)ga;                        // global_addr[31:0]
  g0[3] = (unsigned)((ga >> 32) & 0x01FFFFFFu) // global_addr[56:32]
        | (2u << 30);                          // type = 2 ("image")
  i32x8 g1;
  g1[0] = (3 << 16);                           // data_size = 8 bytes
  g1[1] = (int)((units & 0xFFFFu) << 16);      // tensor_dim0[15:0]
  g1[2] = (int)((units >> 16) & 0xFFFFu)       // tensor_dim0[31:16]
        | (1 << 16);                           // tensor_dim1 = 1
  g1[3] = (int)((units & 0xFFFFu) << 16);      // tile_dim0 = units
  g1[4] = 0;                                   // tile_dim1/2 unused
  g1[5] = (int)units;                          // tensor_dim0_stride[31:0]
  g1[6] = 0;
  g1[7] = 0;
  i32x4 z4 = {0, 0, 0, 0};
  i32x8 z8 = {0, 0, 0, 0, 0, 0, 0, 0};
  __builtin_amdgcn_tensor_load_to_lds(g0, g1, z4, z4, z8, 0);
}
#else
#define HAVE_TDM 0
#endif

// ---------------------------------------------------------------------------
// Kernel 1: fused Q/K/V projections, scattered into shifted-window layout.
// qw,kw: (B, 64, 256, 128) bf16   vT: (B, 64, 128, 256) bf16 (channel-major)
// ---------------------------------------------------------------------------
__global__ void __launch_bounds__(256) k_qkv_window(
    const float* __restrict__ src, const float* __restrict__ tgt,
    const float* __restrict__ Wq, const float* __restrict__ Wk,
    const float* __restrict__ Wv,
    unsigned short* __restrict__ qw, unsigned short* __restrict__ kw,
    unsigned short* __restrict__ vT) {
  extern __shared__ unsigned short lds_wt[]; // 3 * 128*128 bf16, [proj][n][k]
  const int tid = threadIdx.x;
  const int b   = blockIdx.x >> 8;   // 256 token strips per batch
  const int tb  = blockIdx.x & 255;
  const int tok0 = tb * 64;

  for (int i = tid; i < 3 * CC * CC; i += blockDim.x) {
    int p = i / (CC * CC), e = i - p * CC * CC;
    const float* Wp = (p == 0) ? Wq : ((p == 1) ? Wk : Wv);
    int nn = e >> 7, kk = e & 127;
    lds_wt[i] = f2bfbits(Wp[kk * CC + nn]);  // transpose to N-major bf16
  }
  __syncthreads();

  const int wave = tid >> 5, lane = tid & 31, n = lane & 15, h = lane >> 4;
  const float* srcb = src + (size_t)b * LL * CC;
  const float* tgtb = tgt + (size_t)b * LL * CC;

  for (int proj = 0; proj < 3; ++proj) {
    const float* A = (proj == 0) ? srcb : tgtb;
    const unsigned short* wt = lds_wt + proj * CC * CC;
    unsigned short* dst = (proj == 0) ? qw : ((proj == 1) ? kw : vT);
    for (int t = wave; t < 32; t += 8) {   // 4 row-tiles x 8 col-tiles
      int rt = t >> 3, ct = t & 7;
      v8f acc = {};
#pragma unroll
      for (int ks = 0; ks < 4; ++ks)
        acc = wmma_bf16(a_frag_f32(A, CC, tok0 + rt * 16, ks * 32),
                        b_frag_nmajor(wt, CC, ks * 32, ct * 16), acc);
      // scatter into shifted-window layout (roll by -8,-8 then 16x16 windows)
#pragma unroll
      for (int r = 0; r < 8; ++r) {
        int m   = r + h * 8;
        int tok = tok0 + rt * 16 + m;
        int c   = ct * 16 + n;
        int hh = tok >> 7, ww = tok & 127;
        int h2 = (hh + 120) & 127, w2 = (ww + 120) & 127;
        int win = ((h2 >> 4) << 3) | (w2 >> 4);
        int pos = ((h2 & 15) << 4) | (w2 & 15);
        size_t wb = (size_t)b * NWIN + win;
        size_t idx = (proj == 2) ? (wb * CC + c) * WP + pos
                                 : (wb * WP + pos) * CC + c;
        dst[idx] = f2bfbits(acc[r]);
      }
    }
  }
}

// ---------------------------------------------------------------------------
// Kernel 2: shifted-window attention, one workgroup (8 waves) per window.
// LDS: K window 64KB + V^T window 64KB + 8 per-wave P strips 64KB = 192KB.
// K/V^T staged by the Tensor Data Mover (one descriptor each, TENSORcnt).
// ---------------------------------------------------------------------------
__global__ void __launch_bounds__(256) k_window_attn(
    const unsigned short* __restrict__ qw, const unsigned short* __restrict__ kw,
    const unsigned short* __restrict__ vT, const float* __restrict__ mask,
    unsigned short* __restrict__ attn_out) {
  extern __shared__ unsigned char smem[];
  unsigned short* k_lds = (unsigned short*)smem;   // [pos][c]  256x128
  unsigned short* v_lds = k_lds + WP * CC;         // [c][pos]  128x256
  unsigned short* p_lds = v_lds + CC * WP;         // 8 * (16 x 256)

  const int tid = threadIdx.x;
  const int b   = blockIdx.x >> 6;
  const int win = blockIdx.x & 63;
  const size_t wb = (size_t)b * NWIN + win;
  const unsigned short* qg = qw + wb * WP * CC;

#if HAVE_TDM
  // Generic pointers into LDS carry the LDS byte offset in their low 32 bits
  // (ISA 10.2 aperture rules).
  if (tid < 32) {  // wave 0 issues both DMAs (TDM ignores EXEC)
    tdm_load_1d(kw + wb * WP * CC, (unsigned)(size_t)k_lds, WP * CC * 2);
    tdm_load_1d(vT + wb * CC * WP, (unsigned)(size_t)v_lds, CC * WP * 2);
    __builtin_amdgcn_s_wait_tensorcnt(0);
  }
#else
  {
    const u32x4* ksrc = (const u32x4*)(kw + wb * WP * CC);
    const u32x4* vsrc = (const u32x4*)(vT + wb * CC * WP);
    u32x4* kd = (u32x4*)k_lds;
    u32x4* vd = (u32x4*)v_lds;
    for (int i = tid; i < WP * CC / 8; i += blockDim.x) {
      kd[i] = ksrc[i];
      vd[i] = vsrc[i];
    }
  }
#endif
  __syncthreads();

  const int wave = tid >> 5, lane = tid & 31, n = lane & 15, h = lane >> 4;
  unsigned short* pw = p_lds + wave * 16 * WP;
  const float* mwin  = mask + (size_t)win * WP * WP;
  const float rscale = 0.08838834764831845f;  // 1/sqrt(128)

  for (int rr = 0; rr < 2; ++rr) {
    const int row0 = (wave + rr * 8) * 16;
    __builtin_prefetch(qg + (size_t)row0 * CC, 0, 3);

    v16bf aq[4];
#pragma unroll
    for (int ks = 0; ks < 4; ++ks) aq[ks] = a_frag_bf(qg, CC, row0, ks * 32);

    // S = q k^T : 16 column tiles of 16, K = 128
    v8f acc[16];
#pragma unroll
    for (int ct = 0; ct < 16; ++ct) {
      v8f a = {};
#pragma unroll
      for (int ks = 0; ks < 4; ++ks)
        a = wmma_bf16(aq[ks], b_frag_nmajor(k_lds, CC, ks * 32, ct * 16), a);
      acc[ct] = a;
    }

    // scale + mask + softmax over 256 columns (per-lane 16, half-wave 16x)
    float rmax[8], rsum[8];
#pragma unroll
    for (int r = 0; r < 8; ++r) { rmax[r] = -3.0e38f; rsum[r] = 0.f; }
#pragma unroll
    for (int ct = 0; ct < 16; ++ct)
#pragma unroll
      for (int r = 0; r < 8; ++r) {
        int m = r + h * 8;
        float s = acc[ct][r] * rscale +
                  mwin[(size_t)(row0 + m) * WP + ct * 16 + n];
        acc[ct][r] = s;
        rmax[r] = fmaxf(rmax[r], s);
      }
#pragma unroll
    for (int r = 0; r < 8; ++r) rmax[r] = half16_max(rmax[r]);
#pragma unroll
    for (int ct = 0; ct < 16; ++ct)
#pragma unroll
      for (int r = 0; r < 8; ++r) {
        float e = __expf(acc[ct][r] - rmax[r]);
        acc[ct][r] = e;
        rsum[r] += e;
      }
#pragma unroll
    for (int r = 0; r < 8; ++r) rsum[r] = 1.0f / half16_sum(rsum[r]);

    // normalized P (bf16) -> per-wave LDS strip, 16 x 256 row-major
#pragma unroll
    for (int ct = 0; ct < 16; ++ct)
#pragma unroll
      for (int r = 0; r < 8; ++r) {
        int m = r + h * 8;
        pw[m * WP + ct * 16 + n] = f2bfbits(acc[ct][r] * rsum[r]);
      }

    // O = P V : K = 256, 8 channel tiles
    v16bf ap[8];
#pragma unroll
    for (int ks = 0; ks < 8; ++ks) ap[ks] = a_frag_bf(pw, WP, 0, ks * 32);
#pragma unroll
    for (int ct = 0; ct < 8; ++ct) {
      v8f o = {};
#pragma unroll
      for (int ks = 0; ks < 8; ++ks)
        o = wmma_bf16(ap[ks], b_frag_nmajor(v_lds, WP, ks * 32, ct * 16), o);
      // scatter to merged, un-shifted token order (bf16 for the Wm GEMM)
#pragma unroll
      for (int r = 0; r < 8; ++r) {
        int m   = r + h * 8;
        int pos = row0 + m;
        int h2 = ((win >> 3) << 4) | (pos >> 4);
        int w2 = ((win & 7) << 4) | (pos & 15);
        int hh = (h2 + 8) & 127, ww = (w2 + 8) & 127;
        size_t tok = (size_t)hh * WW2 + ww;
        attn_out[((size_t)b * LL + tok) * CC + ct * 16 + n] = f2bfbits(o[r]);
      }
    }
  }
}

// ---------------------------------------------------------------------------
// Kernel 3: message = LN(attn_out @ Wm); ffn_in = [src | message] in bf16.
// ---------------------------------------------------------------------------
__global__ void __launch_bounds__(256) k_msg_ln_concat(
    const unsigned short* __restrict__ attn_out, const float* __restrict__ Wm,
    const float* __restrict__ g1, const float* __restrict__ b1,
    const float* __restrict__ src, unsigned short* __restrict__ ffn_in) {
  extern __shared__ unsigned short wmT[];  // 128x128 N-major bf16
  const int tid  = threadIdx.x;
  const int b    = blockIdx.x >> 7;
  const int tb   = blockIdx.x & 127;
  const int tok0 = tb * 128;

  for (int i = tid; i < CC * CC; i += blockDim.x) {
    int nn = i >> 7, kk = i & 127;
    wmT[i] = f2bfbits(Wm[kk * CC + nn]);
  }
  __syncthreads();

  // copy source (bf16) into first half of the concat rows
  const float* sb = src + ((size_t)b * LL + tok0) * CC;
  unsigned short* fb = ffn_in + ((size_t)b * LL + tok0) * (2 * CC);
  for (int i = tid; i < 128 * CC; i += blockDim.x) {
    int t = i >> 7, c = i & 127;
    fb[(size_t)t * 256 + c] = f2bfbits(sb[(size_t)t * CC + c]);
  }

  const int wave = tid >> 5, lane = tid & 31, n = lane & 15, h = lane >> 4;
  const unsigned short* Ab = attn_out + ((size_t)b * LL + tok0) * CC;
  const int row0 = wave * 16;

  v16bf af[4];
#pragma unroll
  for (int ks = 0; ks < 4; ++ks) af[ks] = a_frag_bf(Ab, CC, row0, ks * 32);
  v8f acc[8] = {};
#pragma unroll
  for (int ct = 0; ct < 8; ++ct)
#pragma unroll
    for (int ks = 0; ks < 4; ++ks)
      acc[ct] = wmma_bf16(af[ks], b_frag_nmajor(wmT, CC, ks * 32, ct * 16),
                          acc[ct]);

  // LayerNorm over the 128 channels of each row
  float mu[8], rstd[8];
#pragma unroll
  for (int r = 0; r < 8; ++r) {
    float s = 0.f;
#pragma unroll
    for (int ct = 0; ct < 8; ++ct) s += acc[ct][r];
    mu[r] = half16_sum(s) * (1.0f / 128.0f);
  }
#pragma unroll
  for (int r = 0; r < 8; ++r) {
    float s = 0.f;
#pragma unroll
    for (int ct = 0; ct < 8; ++ct) {
      float d = acc[ct][r] - mu[r];
      s += d * d;
    }
    rstd[r] = rsqrtf(half16_sum(s) * (1.0f / 128.0f) + 1e-5f);
  }
#pragma unroll
  for (int ct = 0; ct < 8; ++ct)
#pragma unroll
    for (int r = 0; r < 8; ++r) {
      int m = r + h * 8, c = ct * 16 + n;
      float y = (acc[ct][r] - mu[r]) * rstd[r] * g1[c] + b1[c];
      fb[(size_t)(row0 + m) * 256 + 128 + c] = f2bfbits(y);
    }
}

// ---------------------------------------------------------------------------
// Kernel 4: hidden = gelu(ffn_in @ W1), W1 (256,1024). 64 rows x 256 cols/WG.
// ---------------------------------------------------------------------------
__global__ void __launch_bounds__(256) k_ffn1(
    const unsigned short* __restrict__ ffn_in, const float* __restrict__ W1,
    unsigned short* __restrict__ hidden) {
  extern __shared__ unsigned short w1T[];  // 256x256 slice, [n_local][k]
  const int tid  = threadIdx.x;
  const int tok0 = blockIdx.x * 64;  // over B*L rows
  const int n0   = blockIdx.y * 256;

  for (int i = tid; i < 256 * 256; i += blockDim.x) {
    int nl = i >> 8, k = i & 255;
    w1T[i] = f2bfbits(W1[(size_t)k * HID + n0 + nl]);
  }
  __syncthreads();

  const int wave = tid >> 5, lane = tid & 31, n = lane & 15, h = lane >> 4;
  for (int t = wave; t < 64; t += 8) {  // 4 row-tiles x 16 col-tiles
    int rt = t >> 4, ct = t & 15;
    int row = tok0 + rt * 16;
    v8f acc = {};
#pragma unroll
    for (int ks = 0; ks < 8; ++ks)
      acc = wmma_bf16(a_frag_bf(ffn_in, 2 * CC, row, ks * 32),
                      b_frag_nmajor(w1T, 256, ks * 32, ct * 16), acc);
#pragma unroll
    for (int r = 0; r < 8; ++r) {
      int m = r + h * 8;
      float x = acc[r];
      float g = 0.5f * x * (1.0f + erff(x * 0.70710678118654752f));
      hidden[((size_t)(row + m)) * HID + n0 + ct * 16 + n] = f2bfbits(g);
    }
  }
}

// ---------------------------------------------------------------------------
// Kernel 5: out = src + LN(hidden @ W2). 128 rows x 128 cols per workgroup.
// ---------------------------------------------------------------------------
__global__ void __launch_bounds__(256) k_ffn2_ln_add(
    const unsigned short* __restrict__ hidden, const float* __restrict__ W2,
    const float* __restrict__ g2, const float* __restrict__ b2,
    const float* __restrict__ src, float* __restrict__ out) {
  extern __shared__ unsigned short w2T[];  // 128x1024 [n][k] bf16 (256KB)
  const int tid  = threadIdx.x;
  const int tok0 = blockIdx.x * 128;  // over B*L rows

  for (int i = tid; i < CC * HID; i += blockDim.x) {
    int nn = i >> 10, k = i & 1023;
    w2T[i] = f2bfbits(W2[(size_t)k * CC + nn]);
  }
  __syncthreads();

  const int wave = tid >> 5, lane = tid & 31, n = lane & 15, h = lane >> 4;
  const int row0 = tok0 + wave * 16;

  v8f acc[8] = {};
  for (int ks = 0; ks < 32; ++ks) {
    v16bf a = a_frag_bf(hidden, HID, row0, ks * 32);
#pragma unroll
    for (int ct = 0; ct < 8; ++ct)
      acc[ct] = wmma_bf16(a, b_frag_nmajor(w2T, HID, ks * 32, ct * 16),
                          acc[ct]);
  }

  float mu[8], rstd[8];
#pragma unroll
  for (int r = 0; r < 8; ++r) {
    float s = 0.f;
#pragma unroll
    for (int ct = 0; ct < 8; ++ct) s += acc[ct][r];
    mu[r] = half16_sum(s) * (1.0f / 128.0f);
  }
#pragma unroll
  for (int r = 0; r < 8; ++r) {
    float s = 0.f;
#pragma unroll
    for (int ct = 0; ct < 8; ++ct) {
      float d = acc[ct][r] - mu[r];
      s += d * d;
    }
    rstd[r] = rsqrtf(half16_sum(s) * (1.0f / 128.0f) + 1e-5f);
  }
#pragma unroll
  for (int ct = 0; ct < 8; ++ct)
#pragma unroll
    for (int r = 0; r < 8; ++r) {
      int m = r + h * 8, c = ct * 16 + n;
      size_t idx = (size_t)(row0 + m) * CC + c;
      float y = (acc[ct][r] - mu[r]) * rstd[r] * g2[c] + b2[c];
      out[idx] = src[idx] + y;
    }
}

// ---------------------------------------------------------------------------
// Host launcher
// ---------------------------------------------------------------------------
extern "C" void kernel_launch(void* const* d_in, const int* in_sizes, int n_in,
                              void* d_out, int out_size, void* d_ws,
                              size_t ws_size, hipStream_t stream) {
  (void)in_sizes; (void)n_in; (void)out_size; (void)ws_size;
  const float* src  = (const float*)d_in[0];
  const float* tgt  = (const float*)d_in[1];
  const float* Wq   = (const float*)d_in[2];
  const float* Wk   = (const float*)d_in[3];
  const float* Wv   = (const float*)d_in[4];
  const float* Wm   = (const float*)d_in[5];
  const float* g1   = (const float*)d_in[6];
  const float* b1   = (const float*)d_in[7];
  const float* W1   = (const float*)d_in[8];
  const float* W2   = (const float*)d_in[9];
  const float* g2   = (const float*)d_in[10];
  const float* b2   = (const float*)d_in[11];
  const float* mask = (const float*)d_in[12];
  float* out = (float*)d_out;

  // workspace carve-up (bf16 intermediates)
  size_t off = 0;
  auto carve = [&](size_t bytes) {
    void* p = (char*)d_ws + off;
    off += (bytes + 255) & ~(size_t)255;
    return p;
  };
  unsigned short* qw     = (unsigned short*)carve((size_t)BB * NWIN * WP * CC * 2);
  unsigned short* kw     = (unsigned short*)carve((size_t)BB * NWIN * WP * CC * 2);
  unsigned short* vT     = (unsigned short*)carve((size_t)BB * NWIN * CC * WP * 2);
  unsigned short* attn_o = (unsigned short*)carve((size_t)BB * LL * CC * 2);
  unsigned short* ffn_in = (unsigned short*)carve((size_t)BB * LL * 2 * CC * 2);
  unsigned short* hidden = (unsigned short*)carve((size_t)BB * LL * HID * 2);

  const dim3 blk(256);
  k_qkv_window<<<dim3(BB * 256), blk, 3 * CC * CC * 2, stream>>>(
      src, tgt, Wq, Wk, Wv, qw, kw, vT);

  const size_t attn_lds = (size_t)(WP * CC + CC * WP + 8 * 16 * WP) * 2; // 192KB
  k_window_attn<<<dim3(BB * NWIN), blk, attn_lds, stream>>>(
      qw, kw, vT, mask, attn_o);

  k_msg_ln_concat<<<dim3(BB * 128), blk, CC * CC * 2, stream>>>(
      attn_o, Wm, g1, b1, src, ffn_in);

  k_ffn1<<<dim3((BB * LL) / 64, HID / 256), blk, 256 * 256 * 2, stream>>>(
      ffn_in, W1, hidden);

  k_ffn2_ln_add<<<dim3((BB * LL) / 128), blk, CC * HID * 2, stream>>>(
      hidden, W2, g2, b2, src, out);
}
